// LSTM_Attention_50474455662609
// MI455X (gfx1250) — matrix operand order repaired and checked by reading, not compiled, not run on hardware
//
#include <hip/hip_runtime.h>
#include <hip/hip_bf16.h>

#define S_ 1024
#define E_ 512
#define H_ 512
#define AH_ 12
#define L1_ 16
#define V_ 32000
#define T_ 1024
#define G4_ 2048   // 4*H
#define HE_ 1024   // H+E

typedef __attribute__((ext_vector_type(16))) __bf16 v16bf;
typedef __attribute__((ext_vector_type(8)))  float  v8f;

// ---------------------------------------------------------------- zero output
__global__ void zero_out_kernel(float* out, long n) {
  long i = (long)blockIdx.x * blockDim.x + threadIdx.x;
  long stride = (long)gridDim.x * blockDim.x;
  for (; i < n; i += stride) out[i] = 0.0f;
}

// ------------------------------------------------- prep: bf16 convert + fuse
__global__ void prep_kernel(const float* __restrict__ tout,
                            const float* __restrict__ lstm0,
                            const float* __restrict__ Wih,
                            const float* __restrict__ Whh,
                            const float* __restrict__ bih,
                            const float* __restrict__ bhh,
                            __bf16* __restrict__ in_bf,
                            __bf16* __restrict__ wih_bf,
                            __bf16* __restrict__ whhT,
                            float* __restrict__ bias) {
  int stride = gridDim.x * blockDim.x;
  int g = blockIdx.x * blockDim.x + threadIdx.x;
  // teacher-forced input matrix, row 0 replaced by lstm_in0
  for (int i = g; i < T_ * E_; i += stride) {
    int t = i >> 9, e = i & (E_ - 1);
    float v = (t == 0) ? lstm0[e] : tout[i];
    in_bf[i] = (__bf16)v;
  }
  for (int i = g; i < G4_ * E_; i += stride) wih_bf[i] = (__bf16)Wih[i];
  for (int i = g; i < G4_ * E_; i += stride) {
    int n = i >> 9, k = i & (E_ - 1);
    whhT[k * G4_ + n] = (__bf16)Whh[i];    // [K=512][N=2048] for coalesced matvec
  }
  for (int i = g; i < G4_; i += stride) bias[i] = bih[i] + bhh[i];
}

// ------------------------------------------ A1x[s,j] = A1[j, H:] . x[s] (once)
__global__ void a1x_kernel(const float* __restrict__ A1,
                           const float* __restrict__ x,
                           float* __restrict__ a1x) {
  int g = blockIdx.x * blockDim.x + threadIdx.x;
  if (g >= S_ * AH_) return;
  int s = g / AH_, j = g % AH_;
  const float* ar = A1 + j * HE_ + H_;
  const float* xr = x + s * E_;
  float sum = 0.f;
  for (int e = 0; e < E_; ++e) sum += ar[e] * xr[e];
  a1x[s * 16 + j] = sum;
}

// --------------- pre_gates[T,2048] = in @ W_ih^T + b_ih + b_hh (WMMA bf16)
__global__ void __launch_bounds__(256)
ih_gemm_kernel(const __bf16* __restrict__ A,     // [1024][512] row-major
               const __bf16* __restrict__ W,     // [2048][512] row-major (acts as B^T)
               const float* __restrict__ bias,
               float* __restrict__ C) {          // [1024][2048]
  int wave = threadIdx.x >> 5;
  int lane = threadIdx.x & 31;
  int tile = blockIdx.x * 8 + wave;              // 1024 blocks * 8 waves = 8192 tiles
  int mt = tile >> 7, nt = tile & 127;           // 64 x 128 tiles of 16x16
  int m0 = mt * 16, n0 = nt * 16;
  int r = lane & 15, hi = lane >> 4;
  const __bf16* Arow = A + (m0 + r) * E_;        // A: lane r holds row M=r
  const __bf16* Brow = W + (n0 + r) * E_;        // B(k,n)=W[n][k]: lane n holds col N=r
  v8f acc = {};
  for (int kt = 0; kt < E_ / 32; ++kt) {
    int k0 = kt * 32;
    v16bf a, b;
    // A k order used here (contiguous, matches B): lanes0-15 K=0..15, lanes16-31 K=16..31; VGPR v -> K=2v(+1)
#pragma unroll
    for (int v = 0; v < 4; ++v) {
      int ka = k0 + hi * 16 + 2 * v;
      a[2 * v]           = Arow[ka];
      a[2 * v + 1]       = Arow[ka + 1];
      a[2 * (v + 4)]     = Arow[ka + 8];
      a[2 * (v + 4) + 1] = Arow[ka + 9];
    }
    // B k order used here (contiguous, matches A): lanes0-15 K=0..15, lanes16-31 K=16..31; VGPR v -> K=2v(+1)
#pragma unroll
    for (int v = 0; v < 8; ++v) {
      int kb = k0 + hi * 16 + 2 * v;
      b[2 * v]     = Brow[kb];
      b[2 * v + 1] = Brow[kb + 1];
    }
    acc = __builtin_amdgcn_wmma_f32_16x16x32_bf16(false, a, false, b,
                                                  (short)0, acc, false, false);
  }
  // D layout: VGPR j -> M = j + 8*hi, N = lane&15
  int n = n0 + r;
  float bv = bias[n];
#pragma unroll
  for (int j = 0; j < 8; ++j) {
    int m = m0 + j + hi * 8;
    C[m * G4_ + n] = acc[j] + bv;
  }
}

// ---------------------------------- persistent single-WGP sequential decoder
__global__ void __launch_bounds__(1024)
decoder_kernel(const float* __restrict__ x, const float* __restrict__ A1,
               const float* __restrict__ a1, const float* __restrict__ A2,
               const float* __restrict__ a2, const float* __restrict__ a1x,
               const __bf16* __restrict__ whhT, const float* __restrict__ pre_gates,
               const float* __restrict__ P1, const float* __restrict__ p1,
               const float* __restrict__ P2, const float* __restrict__ p2,
               const int* __restrict__ tword, float* __restrict__ out) {
  __shared__ float h[H_], c[H_], ci[H_];
  __shared__ float cipart[1024];
  __shared__ float e_lds[S_];
  __shared__ float gates[G4_];
  __shared__ float red_m[1024], red_s[1024];
  __shared__ float A1hb[AH_];
  __shared__ float zv[L1_];
  __shared__ float ltw_lds, nll_lds;

  int tid = threadIdx.x;
  int lane = tid & 31, wave = tid >> 5;

  if (tid < H_) { h[tid] = 0.f; c[tid] = 0.f; }
  if (tid == 0) nll_lds = 0.f;
  __syncthreads();

  float a2v = a2[0];

  for (int t = 0; t < T_; ++t) {
    // ---- A: A1hb[j] = a1[j] + A1[j,:H] . h   (h-dependent half of attn MLP)
    if (wave < AH_) {
      int j = wave;
      float sum = 0.f;
      for (int k = lane; k < H_; k += 32) sum += A1[j * HE_ + k] * h[k];
      sum += __shfl_down(sum, 16);
      sum += __shfl_down(sum, 8);
      sum += __shfl_down(sum, 4);
      sum += __shfl_down(sum, 2);
      sum += __shfl_down(sum, 1);
      if (lane == 0) A1hb[j] = sum + a1[j];
    }
    __syncthreads();

    // ---- B: raw scores e[s], then block-sum for the raw normalizer
    {
      const float* ax = a1x + tid * 16;
      float ev = a2v;
#pragma unroll
      for (int j = 0; j < AH_; ++j) {
        float hv = A1hb[j] + ax[j];
        hv = (hv > 0.f) ? hv : 0.01f * hv;   // leaky_relu
        ev += A2[j] * hv;
      }
      e_lds[tid] = ev;
      red_s[tid] = ev;
    }
    __syncthreads();
    for (int st = 512; st >= 1; st >>= 1) {
      if (tid < st) red_s[tid] += red_s[tid + st];
      __syncthreads();
    }
    float inv_esum = 1.0f / red_s[0];

    // ---- C: context ci = (1/sum e) * sum_s e[s] * x[s]; split S across 2 halves
    {
      int e = tid & (H_ - 1);
      int s0 = (tid >> 9) * 512;
      float part = 0.f;
      for (int s = s0; s < s0 + 512; ++s) part += e_lds[s] * x[s * E_ + e];
      cipart[tid] = part;
    }
    __syncthreads();
    if (tid < H_) ci[tid] = (cipart[tid] + cipart[tid + 512]) * inv_esum;

    // ---- D: gates = pre_gates[t] + W_hh . h  (old h); coalesced via W_hh^T
    {
      const float* pg = pre_gates + t * G4_;
      float acc0 = pg[tid];
      float acc1 = pg[tid + 1024];
      __builtin_prefetch(pg + G4_ + tid, 0, 1);   // next step's pre_gates row
      for (int k = 0; k < H_; ++k) {
        float hk = h[k];
        acc0 += hk * (float)whhT[k * G4_ + tid];
        acc1 += hk * (float)whhT[k * G4_ + tid + 1024];
      }
      gates[tid] = acc0;
      gates[tid + 1024] = acc1;
    }
    __syncthreads();

    // ---- LSTM elementwise (gate order i,f,g,o)
    if (tid < H_) {
      float ig = gates[tid];
      float fg = gates[H_ + tid];
      float gg = gates[2 * H_ + tid];
      float og = gates[3 * H_ + tid];
      float si = 1.f / (1.f + __expf(-ig));
      float sf = 1.f / (1.f + __expf(-fg));
      float so = 1.f / (1.f + __expf(-og));
      float cn = sf * c[tid] + si * tanhf(gg);
      float hn = so * tanhf(cn);
      c[tid] = cn;
      h[tid] = hn;
    }
    if (tid == 0) ltw_lds = 0.f;
    __syncthreads();

    // ---- E: z[l] = p1[l] + P1[l] . [h_new ; ci]
    if (wave < L1_) {
      int l = wave;
      float sum = 0.f;
      for (int k = lane; k < HE_; k += 32) {
        float v = (k < H_) ? h[k] : ci[k - H_];
        sum += P1[l * HE_ + k] * v;
      }
      sum += __shfl_down(sum, 16);
      sum += __shfl_down(sum, 8);
      sum += __shfl_down(sum, 4);
      sum += __shfl_down(sum, 2);
      sum += __shfl_down(sum, 1);
      if (lane == 0) zv[l] = sum + p1[l];
    }
    __syncthreads();

    // ---- F: V=32000 logits, online (max, sumexp) per thread, then block merge
    {
      int tw = tword[t];
      float zr[L1_];
#pragma unroll
      for (int l = 0; l < L1_; ++l) zr[l] = zv[l];
      float m = -3.402823466e38f, s = 0.f;
      for (int v = tid; v < V_; v += 1024) {
        const float* pr = P2 + v * L1_;
        float lg = p2[v];
#pragma unroll
        for (int l = 0; l < L1_; ++l) lg += pr[l] * zr[l];
        if (lg > m) { s = s * __expf(m - lg) + 1.f; m = lg; }
        else        { s += __expf(lg - m); }
        if (v == tw) ltw_lds = lg;
      }
      red_m[tid] = m;
      red_s[tid] = s;
    }
    __syncthreads();
    for (int st = 512; st >= 1; st >>= 1) {
      if (tid < st) {
        float m1 = red_m[tid],      s1 = red_s[tid];
        float m2 = red_m[tid + st], s2 = red_s[tid + st];
        float M = fmaxf(m1, m2);
        red_m[tid] = M;
        red_s[tid] = s1 * __expf(m1 - M) + s2 * __expf(m2 - M);
      }
      __syncthreads();
    }
    if (tid == 0) {
      int tw = tword[t];
      if (tw >= 0 && tw < V_) {
        float lse = red_m[0] + __logf(red_s[0]);
        nll_lds += lse - ltw_lds;
      }
    }
    __syncthreads();
  }
  if (tid == 0) out[(long)V_ * T_] = nll_lds;   // total_loss (B=1)
}

// ---------------------------------------------------------------- launcher
extern "C" void kernel_launch(void* const* d_in, const int* in_sizes, int n_in,
                              void* d_out, int out_size, void* d_ws, size_t ws_size,
                              hipStream_t stream) {
  (void)in_sizes; (void)n_in; (void)ws_size;
  const float* x    = (const float*)d_in[0];
  const float* tout = (const float*)d_in[1];
  const int*   tw   = (const int*)d_in[2];
  const float* l0   = (const float*)d_in[3];
  const float* Wih  = (const float*)d_in[4];
  const float* Whh  = (const float*)d_in[5];
  const float* bih  = (const float*)d_in[6];
  const float* bhh  = (const float*)d_in[7];
  const float* A1   = (const float*)d_in[8];
  const float* a1   = (const float*)d_in[9];
  const float* A2   = (const float*)d_in[10];
  const float* a2   = (const float*)d_in[11];
  const float* P1   = (const float*)d_in[12];
  const float* p1   = (const float*)d_in[13];
  const float* P2   = (const float*)d_in[14];
  const float* p2   = (const float*)d_in[15];
  float* out = (float*)d_out;

  // workspace layout (~13.7 MB)
  char* w = (char*)d_ws;
  float*  pre_gates = (float*)(w);                  // 1024*2048*4  = 8 MiB
  __bf16* in_bf     = (__bf16*)(w + 8388608);       // 1024*512*2   = 1 MiB
  __bf16* wih_bf    = (__bf16*)(w + 9437184);       // 2048*512*2   = 2 MiB
  __bf16* whhT      = (__bf16*)(w + 11534336);      // 512*2048*2   = 2 MiB
  float*  bias      = (float*)(w + 13631488);       // 2048*4
  float*  a1x       = (float*)(w + 13639680);       // 1024*16*4

  zero_out_kernel<<<4096, 256, 0, stream>>>(out, (long)out_size);
  prep_kernel<<<2048, 256, 0, stream>>>(tout, l0, Wih, Whh, bih, bhh,
                                        in_bf, wih_bf, whhT, bias);
  a1x_kernel<<<(S_ * AH_ + 255) / 256, 256, 0, stream>>>(A1, x, a1x);
  ih_gemm_kernel<<<1024, 256, 0, stream>>>(in_bf, wih_bf, bias, pre_gates);
  decoder_kernel<<<1, 1024, 0, stream>>>(x, A1, a1, A2, a2, a1x, whhT, pre_gates,
                                         P1, p1, P2, p2, tw, out);
}
